// MultiBoxLoss_14181982011619
// MI455X (gfx1250) — compile-verified, matrix operand order checked by reading, not run on hardware
//
#include <hip/hip_runtime.h>
#include <math.h>

#define B_SZ   64
#define P_SZ   24564
#define C_SZ   81
#define NOBJ   16
#define THRESH 0.5f
#define NEG_POS 3
#define TILES_PER_IMG ((P_SZ + 15) / 16)   // 1536
#define SORT_N 32768

typedef __attribute__((ext_vector_type(16))) _Float16 v16h;
typedef __attribute__((ext_vector_type(8)))  float    v8f;

__device__ __forceinline__ float smooth_l1(float d) {
    float ad = fabsf(d);
    return (ad < 1.0f) ? 0.5f * d * d : ad - 0.5f;
}

// ---------------------------------------------------------------------------
// Kernel 1: matching + positives + smooth-L1 localization loss.
// One block per image. Phase 1: per-object argmax over priors (block reduce).
// Phase 2: per-prior argmax (recomputed, cheap ALU), force-match override,
// label assignment, n_pos count, smooth-L1 sum.
// ---------------------------------------------------------------------------
__global__ __launch_bounds__(256) void match_kernel(
    const float* __restrict__ plocs,    // [B,P,4]
    const float* __restrict__ boxes,    // [B,16,4] xy
    const int*   __restrict__ labels,   // [B,16]
    const float* __restrict__ priors,   // [P,4] cxcy
    int*   __restrict__ tc,             // [B,P] true class
    int*   __restrict__ npos,           // [B]
    float* __restrict__ locsum)         // [B]
{
    __shared__ float bx[NOBJ][4];
    __shared__ float a1[NOBJ];
    __shared__ int   lb[NOBJ];
    __shared__ int   ppo[NOBJ];
    __shared__ float sOv[256];
    __shared__ int   sIx[256];
    __shared__ float sF[256];
    __shared__ int   sC[256];

    const int b   = blockIdx.x;
    const int tid = threadIdx.x;

    if (tid < NOBJ) {
        float x0 = boxes[(b * NOBJ + tid) * 4 + 0];
        float y0 = boxes[(b * NOBJ + tid) * 4 + 1];
        float x1 = boxes[(b * NOBJ + tid) * 4 + 2];
        float y1 = boxes[(b * NOBJ + tid) * 4 + 3];
        bx[tid][0] = x0; bx[tid][1] = y0; bx[tid][2] = x1; bx[tid][3] = y1;
        a1[tid] = (x1 - x0) * (y1 - y0);
        lb[tid] = labels[b * NOBJ + tid];
    }
    __syncthreads();

    // ---- Phase 1: per-object best prior (first occurrence on ties) ----
    float bestOv[NOBJ];
    int   bestIx[NOBJ];
#pragma unroll
    for (int o = 0; o < NOBJ; ++o) { bestOv[o] = -1.0f; bestIx[o] = 0x7fffffff; }

    for (int p = tid; p < P_SZ; p += 256) {
        float4 pc = ((const float4*)priors)[p];
        float px0 = pc.x - pc.z * 0.5f, py0 = pc.y - pc.w * 0.5f;
        float px1 = pc.x + pc.z * 0.5f, py1 = pc.y + pc.w * 0.5f;
        float a2  = (px1 - px0) * (py1 - py0);
#pragma unroll
        for (int o = 0; o < NOBJ; ++o) {
            float lx = fmaxf(bx[o][0], px0), ly = fmaxf(bx[o][1], py0);
            float hx = fminf(bx[o][2], px1), hy = fminf(bx[o][3], py1);
            float inter = fmaxf(hx - lx, 0.0f) * fmaxf(hy - ly, 0.0f);
            float ov = inter / (a1[o] + a2 - inter);
            if (ov > bestOv[o]) { bestOv[o] = ov; bestIx[o] = p; }
        }
    }

#pragma unroll 1
    for (int o = 0; o < NOBJ; ++o) {
        sOv[tid] = bestOv[o]; sIx[tid] = bestIx[o];
        __syncthreads();
        for (int s = 128; s > 0; s >>= 1) {
            if (tid < s) {
                float ov2 = sOv[tid + s]; int ix2 = sIx[tid + s];
                if (ov2 > sOv[tid] || (ov2 == sOv[tid] && ix2 < sIx[tid])) {
                    sOv[tid] = ov2; sIx[tid] = ix2;
                }
            }
            __syncthreads();
        }
        if (tid == 0) ppo[o] = sIx[0];
        __syncthreads();
    }

    // ---- Phase 2: per-prior assignment + loc loss ----
    int   myNpos = 0;
    float myLoc  = 0.0f;
    for (int p = tid; p < P_SZ; p += 256) {
        float4 pc = ((const float4*)priors)[p];
        float px0 = pc.x - pc.z * 0.5f, py0 = pc.y - pc.w * 0.5f;
        float px1 = pc.x + pc.z * 0.5f, py1 = pc.y + pc.w * 0.5f;
        float a2  = (px1 - px0) * (py1 - py0);
        float maxOv = -INFINITY; int argO = 0;
#pragma unroll
        for (int o = 0; o < NOBJ; ++o) {
            float lx = fmaxf(bx[o][0], px0), ly = fmaxf(bx[o][1], py0);
            float hx = fminf(bx[o][2], px1), hy = fminf(bx[o][3], py1);
            float inter = fmaxf(hx - lx, 0.0f) * fmaxf(hy - ly, 0.0f);
            float ov = inter / (a1[o] + a2 - inter);
            if (ov > maxOv) { maxOv = ov; argO = o; }
        }
        // force-match override (later object wins, matching sequential scatter)
#pragma unroll
        for (int o = 0; o < NOBJ; ++o)
            if (p == ppo[o]) { argO = o; maxOv = 1.0f; }

        int lab = (maxOv < THRESH) ? 0 : lb[argO];
        tc[(long long)b * P_SZ + p] = lab;
        if (lab != 0) {
            ++myNpos;
            float x0 = bx[argO][0], y0 = bx[argO][1];
            float x1 = bx[argO][2], y1 = bx[argO][3];
            float cx = (x0 + x1) * 0.5f, cy = (y0 + y1) * 0.5f;
            float w = x1 - x0, h = y1 - y0;
            float g0 = (cx - pc.x) / (pc.z / 10.0f);
            float g1 = (cy - pc.y) / (pc.w / 10.0f);
            float g2 = logf(w / pc.z) * 5.0f;
            float g3 = logf(h / pc.w) * 5.0f;
            float4 pl = ((const float4*)plocs)[(long long)b * P_SZ + p];
            myLoc += smooth_l1(pl.x - g0) + smooth_l1(pl.y - g1) +
                     smooth_l1(pl.z - g2) + smooth_l1(pl.w - g3);
        }
    }

    sF[tid] = myLoc; sC[tid] = myNpos;
    __syncthreads();
    for (int s = 128; s > 0; s >>= 1) {
        if (tid < s) { sF[tid] += sF[tid + s]; sC[tid] += sC[tid + s]; }
        __syncthreads();
    }
    if (tid == 0) { npos[b] = sC[0]; locsum[b] = sF[0]; }
}

// ---------------------------------------------------------------------------
// Kernel 2: per-prior cross-entropy (bandwidth-dominant, 509 MB stream).
// One wave32 per 16-prior tile. Lane layout matches the CDNA5 f16 A-matrix
// 16x32 layout; Sum(exp) over 81 classes is computed with three chained
// v_wmma_f32_16x16x32_f16 ops against an all-ones B matrix (f32 accum).
// Row sums are read straight out of the D-matrix VGPRs of the owning lanes
// (no cross-lane traffic); the bulk stream uses non-temporal loads.
// ---------------------------------------------------------------------------
__global__ __launch_bounds__(256) void ce_kernel(
    const float* __restrict__ scores,   // [B,P,81]
    const int*   __restrict__ tc,       // [B,P]
    float* __restrict__ ceneg,          // [B,P]
    float* __restrict__ cepos_part)     // [gridDim.x]
{
    const int tid  = threadIdx.x;
    const int lane = tid & 31;
    const int wid  = (blockIdx.x * 256 + tid) >> 5;
    const int b    = wid / TILES_PER_IMG;
    const int tile = wid % TILES_PER_IMG;
    const int row  = lane & 15;
    const int half = lane >> 4;
    // clamp tail rows onto valid memory: their WMMA rows are computed but
    // never written, so duplicated data is harmless and avoids divergence.
    int p = tile * 16 + row;
    if (p >= P_SZ) p = P_SZ - 1;
    const long long rowbase = ((long long)b * P_SZ + p) * C_SZ;
    const float* rp = scores + rowbase + half * 8;

    // prefetch next tile's stream (global_prefetch_b8)
    __builtin_prefetch(scores + rowbase + 16 * C_SZ, 0, 1);

    // ---- pass 1: unconditional vectorizable loads + running max ----
    // A-matrix class offsets for this lane: lane<16 -> {0-7,16-23},
    // lane>=16 -> {8-15,24-31} of each 32-class chunk.
    float raw[48];
#pragma unroll
    for (int c = 0; c < 2; ++c) {
#pragma unroll
        for (int j = 0; j < 8; ++j)
            raw[c * 16 + j]     = __builtin_nontemporal_load(rp + c * 32 + j);
#pragma unroll
        for (int j = 0; j < 8; ++j)
            raw[c * 16 + 8 + j] = __builtin_nontemporal_load(rp + c * 32 + 16 + j);
    }
    // chunk 2 low half: classes 64+half*8+{0..7}, all < 81
#pragma unroll
    for (int j = 0; j < 8; ++j)
        raw[32 + j] = __builtin_nontemporal_load(rp + 64 + j);
    // chunk 2 high half: classes 80..95 -> only class 80 (half==0,j==0) valid
    raw[40] = (half == 0) ? scores[rowbase + 80] : -INFINITY;
#pragma unroll
    for (int j = 1; j < 8; ++j) raw[40 + j] = -INFINITY;

    float m = -INFINITY;
#pragma unroll
    for (int i = 0; i < 48; ++i) m = fmaxf(m, raw[i]);
    // row max: the two half-lanes of each row exchange (always finite)
    m = fmaxf(m, __shfl_xor(m, 16));

    // ---- pass 2: exp in f16, WMMA row-sum against all-ones B ----
    v16h ones;
#pragma unroll
    for (int i = 0; i < 16; ++i) ones[i] = (_Float16)1.0f;

    v8f acc = {};
#pragma unroll
    for (int c = 0; c < 3; ++c) {
        v16h a;
#pragma unroll
        for (int j = 0; j < 16; ++j)
            a[j] = (_Float16)__expf(raw[c * 16 + j] - m);
        acc = __builtin_amdgcn_wmma_f32_16x16x32_f16(
            false, a, false, ones, (short)0, acc, false, false);
    }

    // ---- row-sum extraction without cross-lane ops ----
    // D layout: lane L<16 holds rows 0-7 (col L) in acc[0..7];
    //           lane 16+t holds rows 8-15 (col t) in acc[0..7].
    // Writers: lanes 0-7 -> rows 0-7 (own acc[lane]); lanes 16-23 -> rows 8-15.
    float mysum = 0.0f;
#pragma unroll
    for (int r = 0; r < 8; ++r)
        if ((lane & 7) == r) mysum = acc[r];

    // high writers (lane 16+t) need the row max of row 8+t, held by lane 8+t
    float m_other = __shfl_xor(m, 24);
    const bool writer_hi = (lane >= 16) && (lane < 24);
    const bool is_writer = (lane < 8) || writer_hi;
    float m_eff = writer_hi ? m_other : m;

    float myce = 0.0f;
    if (is_writer) {
        int wrow = (lane & 7) + (writer_hi ? 8 : 0);
        int wp = tile * 16 + wrow;
        if (wp < P_SZ) {
            long long widx = (long long)b * P_SZ + wp;
            int lab   = tc[widx];
            float xt  = scores[widx * C_SZ + lab];
            float ce  = m_eff + __logf(mysum) - xt;   // -log_softmax[true]
            bool pos  = (lab != 0);
            ceneg[widx] = pos ? 0.0f : ce;
            if (pos) myce = ce;
        }
    }

    __shared__ float red[256];
    red[tid] = myce;
    __syncthreads();
    for (int s = 128; s > 0; s >>= 1) {
        if (tid < s) red[tid] += red[tid + s];
        __syncthreads();
    }
    if (tid == 0) cepos_part[blockIdx.x] = red[0];
}

// ---------------------------------------------------------------------------
// Kernel 3: hard-negative mining. Bitonic sort (descending) of one image's
// negative CE values in 128 KB of dynamic LDS, then sum of top 3*n_pos.
// Padding with 0.0 is sum-equivalent (all real entries are >= 0).
// ---------------------------------------------------------------------------
__global__ __launch_bounds__(1024) void hardneg_kernel(
    const float* __restrict__ ceneg,
    const int*   __restrict__ npos,
    float* __restrict__ hard)
{
    extern __shared__ float s[];
    const int b = blockIdx.x, tid = threadIdx.x;

    for (int i = tid; i < SORT_N; i += 1024)
        s[i] = (i < P_SZ) ? ceneg[(long long)b * P_SZ + i] : 0.0f;
    __syncthreads();

    for (int k2 = 2; k2 <= SORT_N; k2 <<= 1) {
        for (int j = k2 >> 1; j > 0; j >>= 1) {
            for (int i = tid; i < SORT_N; i += 1024) {
                int ixj = i ^ j;
                if (ixj > i) {
                    float a = s[i], c = s[ixj];
                    bool descSeg = ((i & k2) == 0);
                    bool doSwap = descSeg ? (a < c) : (a > c);
                    if (doSwap) { s[i] = c; s[ixj] = a; }
                }
            }
            __syncthreads();
        }
    }

    int k = NEG_POS * npos[b];
    if (k > SORT_N) k = SORT_N;
    float local = 0.0f;
    for (int i = tid; i < k; i += 1024) local += s[i];
    __syncthreads();
    s[tid] = local;
    __syncthreads();
    for (int st = 512; st > 0; st >>= 1) {
        if (tid < st) s[tid] += s[tid + st];
        __syncthreads();
    }
    if (tid == 0) hard[b] = s[0];
}

// ---------------------------------------------------------------------------
// Kernel 4: deterministic final reduction to the scalar loss.
// ---------------------------------------------------------------------------
__global__ __launch_bounds__(256) void final_kernel(
    const int*   __restrict__ npos,
    const float* __restrict__ locsum,
    const float* __restrict__ cepos_part, int nparts,
    const float* __restrict__ hard,
    float* __restrict__ out)
{
    __shared__ float rA[256];
    __shared__ float rB[256];
    __shared__ int   rI[256];
    const int tid = threadIdx.x;

    float ce = 0.0f;
    for (int i = tid; i < nparts; i += 256) ce += cepos_part[i];
    float hd = 0.0f, lc = 0.0f; int np = 0;
    for (int i = tid; i < B_SZ; i += 256) {
        hd += hard[i]; lc += locsum[i]; np += npos[i];
    }
    rA[tid] = ce + hd; rB[tid] = lc; rI[tid] = np;
    __syncthreads();
    for (int s = 128; s > 0; s >>= 1) {
        if (tid < s) { rA[tid] += rA[tid + s]; rB[tid] += rB[tid + s]; rI[tid] += rI[tid + s]; }
        __syncthreads();
    }
    if (tid == 0) {
        float tp = (float)rI[0];
        out[0] = rA[0] / tp + rB[0] / (tp * 4.0f);
    }
}

// ---------------------------------------------------------------------------
extern "C" void kernel_launch(void* const* d_in, const int* in_sizes, int n_in,
                              void* d_out, int out_size, void* d_ws, size_t ws_size,
                              hipStream_t stream)
{
    const float* plocs  = (const float*)d_in[0];   // [B,P,4]
    const float* scores = (const float*)d_in[1];   // [B,P,81]
    const float* boxes  = (const float*)d_in[2];   // [B,16,4]
    const int*   labels = (const int*)d_in[3];     // [B,16]
    const float* priors = (const float*)d_in[4];   // [P,4]
    float* out = (float*)d_out;

    const int CE_BLOCKS = (B_SZ * TILES_PER_IMG * 32) / 256;  // 12288

    char* ws = (char*)d_ws;
    size_t off = 0;
    int*   tc     = (int*)(ws + off);   off += (size_t)B_SZ * P_SZ * sizeof(int);
    float* ceneg  = (float*)(ws + off); off += (size_t)B_SZ * P_SZ * sizeof(float);
    int*   npos   = (int*)(ws + off);   off += (size_t)B_SZ * sizeof(int);
    float* locsum = (float*)(ws + off); off += (size_t)B_SZ * sizeof(float);
    float* hard   = (float*)(ws + off); off += (size_t)B_SZ * sizeof(float);
    float* cepos  = (float*)(ws + off); off += (size_t)CE_BLOCKS * sizeof(float);

    match_kernel<<<B_SZ, 256, 0, stream>>>(plocs, boxes, labels, priors,
                                           tc, npos, locsum);
    ce_kernel<<<CE_BLOCKS, 256, 0, stream>>>(scores, tc, ceneg, cepos);
    hardneg_kernel<<<B_SZ, 1024, SORT_N * sizeof(float), stream>>>(ceneg, npos, hard);
    final_kernel<<<1, 256, 0, stream>>>(npos, locsum, cepos, CE_BLOCKS, hard, out);
}
